// TNetkd_2877628088368
// MI455X (gfx1250) — compile-verified, matrix operand order checked by reading, not run on hardware
//
#include <hip/hip_runtime.h>
#include <hip/hip_bf16.h>
#include <math.h>

#define KNN 32
#define BN_EPS 1e-5f
#define NPTS 4096
#define MTOT 16384   // bs * NPTS

typedef __attribute__((ext_vector_type(16))) __bf16 v16bf;
typedef __attribute__((ext_vector_type(8)))  float  v8f;

// ---------------------------------------------------------------------------
// Layer 1: 3 -> 64 conv1x1 (Cin=3 too small for WMMA; pure streaming VALU)
// x layout: [bs][3][4096]; Y1 layout: [64][MTOT], col j = b*4096 + p
// ---------------------------------------------------------------------------
__global__ void conv1_kernel(const float* __restrict__ x,
                             const float* __restrict__ W1,
                             float* __restrict__ Y1)
{
    int j = blockIdx.x * blockDim.x + threadIdx.x;
    if (j >= MTOT) return;
    int b = j >> 12, p = j & (NPTS - 1);
    const float* xb = x + (size_t)b * 3 * NPTS;
    float p0 = xb[p], p1 = xb[NPTS + p], p2 = xb[2 * NPTS + p];
    #pragma unroll 4
    for (int o = 0; o < 64; ++o) {
        Y1[(size_t)o * MTOT + j] =
            W1[o * 3 + 0] * p0 + W1[o * 3 + 1] * p1 + W1[o * 3 + 2] * p2;
    }
}

// ---------------------------------------------------------------------------
// KNN(32) + Gaussian kernel conv (3 -> 128).  One thread per point.
// Writes kern_map rows into X3 channels [128..255]: Xk[o][b*4096+p].
// ---------------------------------------------------------------------------
__global__ void knn_gauss_kernel(const float* __restrict__ x,
                                 const float* __restrict__ kcW,
                                 const float* __restrict__ kcb,
                                 float* __restrict__ Xk)
{
    int j = blockIdx.x * blockDim.x + threadIdx.x;
    if (j >= MTOT) return;
    int b = j >> 12, i = j & (NPTS - 1);
    const float* xb = x + (size_t)b * 3 * NPTS;
    float px = xb[i], py = xb[NPTS + i], pz = xb[2 * NPTS + i];

    float nn_d[KNN];
    int   nn_i[KNN];
    #pragma unroll
    for (int k = 0; k < KNN; ++k) { nn_d[k] = 3.4e38f; nn_i[k] = 0; }

    for (int q = 0; q < NPTS; ++q) {
        float dx = px - xb[q];
        float dy = py - xb[NPTS + q];
        float dz = pz - xb[2 * NPTS + q];
        float d2 = dx * dx + dy * dy + dz * dz;
        if (d2 < nn_d[KNN - 1]) {               // insertion keeps ties stable (low idx first)
            int k = KNN - 1;
            while (k > 0 && nn_d[k - 1] > d2) {
                nn_d[k] = nn_d[k - 1];
                nn_i[k] = nn_i[k - 1];
                --k;
            }
            nn_d[k] = d2; nn_i[k] = q;
        }
    }

    // center = nearest neighbor (== self); Gaussian-weighted aggregate (sigma=1)
    int c0 = nn_i[0];
    float cx = xb[c0], cy = xb[NPTS + c0], cz = xb[2 * NPTS + c0];
    float wsum = 0.f, ax = 0.f, ay = 0.f, az = 0.f;
    #pragma unroll
    for (int k = 0; k < KNN; ++k) {
        int q = nn_i[k];
        float qx = xb[q], qy = xb[NPTS + q], qz = xb[2 * NPTS + q];
        float dx = qx - cx, dy = qy - cy, dz = qz - cz;
        float w = __expf(-0.5f * (dx * dx + dy * dy + dz * dz));
        wsum += w; ax += w * qx; ay += w * qy; az += w * qz;
    }
    float inv = 1.f / wsum;
    ax *= inv; ay *= inv; az *= inv;

    #pragma unroll 4
    for (int o = 0; o < 128; ++o) {
        Xk[(size_t)o * MTOT + j] =
            kcW[o * 3 + 0] * ax + kcW[o * 3 + 1] * ay + kcW[o * 3 + 2] * az + kcb[o];
    }
}

// ---------------------------------------------------------------------------
// Per-channel BN statistics over MTOT positions -> fused affine (scale, shift)
// plus ReLU floor (0). One block per channel.
// ---------------------------------------------------------------------------
__global__ void stats_kernel(const float* __restrict__ Y,
                             const float* __restrict__ g,
                             const float* __restrict__ be,
                             float* __restrict__ scale,
                             float* __restrict__ shift,
                             float* __restrict__ floorv)
{
    int c = blockIdx.x;
    const float* row = Y + (size_t)c * MTOT;
    float s = 0.f, s2 = 0.f;
    for (int j = threadIdx.x; j < MTOT; j += blockDim.x) {
        float v = row[j];
        s += v; s2 += v * v;
    }
    __shared__ float ls[256], lq[256];
    ls[threadIdx.x] = s; lq[threadIdx.x] = s2;
    __syncthreads();
    for (int st = blockDim.x >> 1; st > 0; st >>= 1) {
        if ((int)threadIdx.x < st) {
            ls[threadIdx.x] += ls[threadIdx.x + st];
            lq[threadIdx.x] += lq[threadIdx.x + st];
        }
        __syncthreads();
    }
    if (threadIdx.x == 0) {
        float mean = ls[0] * (1.f / MTOT);
        float var  = lq[0] * (1.f / MTOT) - mean * mean;
        float sc   = g[c] * rsqrtf(var + BN_EPS);
        scale[c]  = sc;
        shift[c]  = be[c] - mean * sc;
        floorv[c] = 0.f;
    }
}

// Identity affine (no BN, no ReLU) for the concatenated kernel-map channels.
__global__ void identity_affine_kernel(float* __restrict__ aff, int C, int off, int cnt)
{
    int t = blockIdx.x * blockDim.x + threadIdx.x;
    if (t >= cnt) return;
    aff[off + t]         = 1.f;
    aff[C + off + t]     = 0.f;
    aff[2 * C + off + t] = -3.4e38f;   // floor = -inf => ReLU disabled
}

// ---------------------------------------------------------------------------
// WMMA GEMM: Y[Cout][M] = W[Cout][Cin] (f32->bf16) x BN_ReLU(X[Cin][M]) (f32->bf16)
// f32 accumulation via v_wmma_f32_16x16x32_bf16.
// Each wave computes a 64x16 tile (4 Cout sub-tiles sharing one B tile):
// the expensive affine-transformed activation tile is loaded/converted once
// and consumed by 4 WMMAs; weight tiles are cheap contiguous b128 loads.
// Requires: Cin % 32 == 0, Cout % 64 == 0 (true for all layers here).
// ---------------------------------------------------------------------------
__global__ void gemm_bf16_wmma(const float* __restrict__ W,
                               const float* __restrict__ X,
                               float* __restrict__ Y,
                               const float* __restrict__ aff, // [scale|shift|floor] x Cin, or null
                               int Cout, int Cin, int M)
{
    const int lane  = threadIdx.x & 31;
    const int wave  = threadIdx.x >> 5;                // 0..3
    const int tileN = (blockIdx.x * 4 + wave) * 16;    // column base (M)
    const int tileM = blockIdx.y * 64;                 // row base (Cout), 4 sub-tiles
    const int half  = lane >> 4;                       // K-half select
    const int l15   = lane & 15;

    v8f acc[4] = {};
    for (int k0 = 0; k0 < Cin; k0 += 32) {
        // --- B tile (shared by all 4 Cout sub-tiles): X columns, fused
        //     per-channel affine + ReLU-floor, f32 -> bf16.
        // elements 0..7  -> K = k0 + half*8 + e
        // elements 8..15 -> K = k0 + 16 + half*8 + (e-8)
        v16bf bm;
        const float* xcol = X + (size_t)(k0 + half * 8) * M + (tileN + l15);
        if (k0 + 32 < Cin) {   // prefetch next K block's activation rows
            __builtin_prefetch((const void*)(xcol + (size_t)32 * M), 0, 0);
            __builtin_prefetch((const void*)(xcol + (size_t)48 * M), 0, 0);
        }
        if (aff) {
            const float* sc = aff;
            const float* sh = aff + Cin;
            const float* fl = aff + 2 * Cin;
            #pragma unroll
            for (int e = 0; e < 8; ++e) {
                int k1 = k0 + half * 8 + e;
                int k2 = k1 + 16;
                float v1 = fmaxf(xcol[(size_t)e * M]        * sc[k1] + sh[k1], fl[k1]);
                float v2 = fmaxf(xcol[(size_t)(16 + e) * M] * sc[k2] + sh[k2], fl[k2]);
                bm[e]     = (__bf16)v1;
                bm[e + 8] = (__bf16)v2;
            }
        } else {
            #pragma unroll
            for (int e = 0; e < 8; ++e) {
                bm[e]     = (__bf16)xcol[(size_t)e * M];
                bm[e + 8] = (__bf16)xcol[(size_t)(16 + e) * M];
            }
        }

        // --- 4 A tiles (weight rows, contiguous -> b128 loads) + 4 WMMAs
        #pragma unroll
        for (int t = 0; t < 4; ++t) {
            v16bf a;
            const float* wrow =
                W + (size_t)(tileM + t * 16 + l15) * Cin + k0 + half * 8;
            #pragma unroll
            for (int e = 0; e < 8; ++e) {
                a[e]     = (__bf16)wrow[e];
                a[e + 8] = (__bf16)wrow[16 + e];
            }
            acc[t] = __builtin_amdgcn_wmma_f32_16x16x32_bf16(
                false, a, false, bm, (short)0, acc[t], false, false);
        }
    }

    // D layout: row = half*8 + r, col = l15
    #pragma unroll
    for (int t = 0; t < 4; ++t) {
        float* yout = Y + (size_t)(tileM + t * 16 + half * 8) * M + tileN + l15;
        #pragma unroll
        for (int r = 0; r < 8; ++r) {
            yout[(size_t)r * M] = acc[t][r];
        }
    }
}

// ---------------------------------------------------------------------------
// Final layer: 256 -> 9 conv + b6, global max over 4096 positions per (b,o),
// + identity. One block per (b,o) pair (36 blocks). BN+ReLU of h5 fused.
// ---------------------------------------------------------------------------
__global__ void final_kernel(const float* __restrict__ Y5,
                             const float* __restrict__ aff5, // scale[256], shift[256]
                             const float* __restrict__ W6,
                             const float* __restrict__ b6,
                             float* __restrict__ out)
{
    int b = blockIdx.x / 9;
    int o = blockIdx.x % 9;
    const float* w = W6 + o * 256;
    float best = -3.4e38f;
    for (int p = threadIdx.x; p < NPTS; p += blockDim.x) {
        size_t col = (size_t)b * NPTS + p;
        float accv = 0.f;
        #pragma unroll 4
        for (int c = 0; c < 256; ++c) {
            float xn = fmaxf(Y5[(size_t)c * MTOT + col] * aff5[c] + aff5[256 + c], 0.f);
            accv += w[c] * xn;
        }
        best = fmaxf(best, accv);
    }
    __shared__ float ls[256];
    ls[threadIdx.x] = best;
    __syncthreads();
    for (int st = blockDim.x >> 1; st > 0; st >>= 1) {
        if ((int)threadIdx.x < st)
            ls[threadIdx.x] = fmaxf(ls[threadIdx.x], ls[threadIdx.x + st]);
        __syncthreads();
    }
    if (threadIdx.x == 0) {
        float iden = (o == 0 || o == 4 || o == 8) ? 1.f : 0.f;
        out[b * 9 + o] = ls[0] + b6[o] + iden;
    }
}

// ---------------------------------------------------------------------------
extern "C" void kernel_launch(void* const* d_in, const int* in_sizes, int n_in,
                              void* d_out, int out_size, void* d_ws, size_t ws_size,
                              hipStream_t stream)
{
    const float* x   = (const float*)d_in[0];
    const float* W1  = (const float*)d_in[1];
    const float* W2  = (const float*)d_in[2];
    const float* W3  = (const float*)d_in[3];
    const float* W4  = (const float*)d_in[4];
    const float* W5  = (const float*)d_in[5];
    const float* W6  = (const float*)d_in[6];
    const float* b6  = (const float*)d_in[7];
    const float* g1  = (const float*)d_in[8];
    const float* be1 = (const float*)d_in[9];
    const float* g2  = (const float*)d_in[10];
    const float* be2 = (const float*)d_in[11];
    const float* g3  = (const float*)d_in[12];
    const float* be3 = (const float*)d_in[13];
    const float* g4  = (const float*)d_in[14];
    const float* be4 = (const float*)d_in[15];
    const float* g5  = (const float*)d_in[16];
    const float* be5 = (const float*)d_in[17];
    const float* kcW = (const float*)d_in[18];
    const float* kcb = (const float*)d_in[19];
    float* out = (float*)d_out;

    // Workspace layout (floats)
    float* ws = (float*)d_ws;
    float* Y1 = ws;                             //   64 * MTOT
    float* X3 = Y1 + (size_t)64 * MTOT;         //  256 * MTOT (rows 0..127 = Y2)
    float* Y3 = X3 + (size_t)256 * MTOT;        // 1024 * MTOT
    float* Y4 = Y3 + (size_t)1024 * MTOT;       //  512 * MTOT
    float* Y5 = Y4 + (size_t)512 * MTOT;        //  256 * MTOT
    float* A1 = Y5 + (size_t)256 * MTOT;        // 3 *   64
    float* A2 = A1 + 3 * 64;                    // 3 *  256
    float* A3 = A2 + 3 * 256;                   // 3 * 1024
    float* A4 = A3 + 3 * 1024;                  // 3 *  512
    float* A5 = A4 + 3 * 512;                   // 3 *  256

    dim3 b256(256);

    // Stage 0: layer-1 conv and KNN/Gaussian kernel map (independent)
    conv1_kernel<<<MTOT / 256, b256, 0, stream>>>(x, W1, Y1);
    knn_gauss_kernel<<<MTOT / 256, b256, 0, stream>>>(x, kcW, kcb, X3 + (size_t)128 * MTOT);

    // BN1 stats; layer 2 (64 -> 128) writes into X3 rows 0..127
    stats_kernel<<<64, b256, 0, stream>>>(Y1, g1, be1, A1, A1 + 64, A1 + 128);
    gemm_bf16_wmma<<<dim3(MTOT / 64, 128 / 64), dim3(128), 0, stream>>>(
        W2, Y1, X3, A1, 128, 64, MTOT);

    // BN2 stats on X3[0..127]; identity affine for kernel-map channels 128..255
    stats_kernel<<<128, b256, 0, stream>>>(X3, g2, be2, A2, A2 + 256, A2 + 512);
    identity_affine_kernel<<<1, 128, 0, stream>>>(A2, 256, 128, 128);

    // Layer 3: 256 -> 1024
    gemm_bf16_wmma<<<dim3(MTOT / 64, 1024 / 64), dim3(128), 0, stream>>>(
        W3, X3, Y3, A2, 1024, 256, MTOT);
    stats_kernel<<<1024, b256, 0, stream>>>(Y3, g3, be3, A3, A3 + 1024, A3 + 2048);

    // Layer 4: 1024 -> 512
    gemm_bf16_wmma<<<dim3(MTOT / 64, 512 / 64), dim3(128), 0, stream>>>(
        W4, Y3, Y4, A3, 512, 1024, MTOT);
    stats_kernel<<<512, b256, 0, stream>>>(Y4, g4, be4, A4, A4 + 512, A4 + 1024);

    // Layer 5: 512 -> 256
    gemm_bf16_wmma<<<dim3(MTOT / 64, 256 / 64), dim3(128), 0, stream>>>(
        W5, Y4, Y5, A4, 256, 512, MTOT);
    stats_kernel<<<256, b256, 0, stream>>>(Y5, g5, be5, A5, A5 + 256, A5 + 512);

    // Final: 256 -> 9, bias, global max, + identity
    final_kernel<<<36, b256, 0, stream>>>(Y5, A5, W6, b6, out);

    (void)in_sizes; (void)n_in; (void)out_size; (void)ws_size;
}
